// GATConvGNNLayer_34772055229052
// MI455X (gfx1250) — compile-verified, matrix operand order checked by reading, not run on hardware
//
#include <hip/hip_runtime.h>
#include <hip/hip_bf16.h>
#include <stdint.h>

#define N_NODES 100000
#define N_EDGES 1600000
#define DIM     128
#define HEADS   4
#define D_HEAD  32
#define NEG_SLOPE 0.2f
#define BN_EPS    1e-5f

typedef __attribute__((ext_vector_type(16))) _Float16 v16h;
typedef __attribute__((ext_vector_type(8)))  float    v8f;
typedef __attribute__((ext_vector_type(4)))  int      v4i;

// gfx1250 async global->LDS staging (GLOBAL_LOAD_ASYNC_TO_LDS_B128), guarded
// so the file compiles on toolchains without the builtin.  Probe-confirmed:
// the builtin exists here and takes (int4 AS1*, int4 AS3*, imm offset, cpol).
#if defined(__has_builtin)
#  if __has_builtin(__builtin_amdgcn_global_load_async_to_lds_b128)
#    define USE_ASYNC_LDS 1
#  endif
#endif

#if defined(USE_ASYNC_LDS)
typedef __attribute__((address_space(1))) void* as1_vptr;
typedef __attribute__((address_space(3))) void* as3_vptr;
typedef __attribute__((address_space(1))) v4i*  as1_v4i;
typedef __attribute__((address_space(3))) v4i*  as3_v4i;
__device__ __forceinline__ void async_wait0() {
#if __has_builtin(__builtin_amdgcn_s_wait_asynccnt)
    __builtin_amdgcn_s_wait_asynccnt(0);
#else
    asm volatile("s_wait_asynccnt 0" ::: "memory");
#endif
}
#endif

// ---------------------------------------------------------------------------
// GEMM C[M x 128] = A[M x 128] * B[128 x 128], fp32 in/out, f16 WMMA compute.
// One block = 16-row strip; 8 waves, wave w owns the 16x16 tile at cols w*16.
// The 16x128 A strip is staged once in LDS (async-to-LDS when available)
// instead of being re-read 8x (once per wave) from global.
// Fragment layouts per CDNA5 ISA 7.12.2 (wave32):
//   A (16x32 f16): lane<16 -> row=lane&15, K groups {0..7,16..23}; lane>=16 ->
//   K groups {8..15,24..31}.  B mirrors with col=lane&15.
//   C/D (16x16 f32): vgpr r holds row r+8*(lane>>4), col lane&15.
// ---------------------------------------------------------------------------
#define LDA 132   // padded row stride (floats); 132 % 64 == 4 -> spread banks
__global__ __launch_bounds__(256) void gemm128_wmma(const float* __restrict__ A,
                                                    const float* __restrict__ B,
                                                    float* __restrict__ C) {
    __shared__ float lA[16][LDA];

    // ---- stage the 16x128 fp32 A strip into LDS -------------------------
    {
        const int r = threadIdx.x >> 4;          // 0..15  (row of strip)
        const int c = threadIdx.x & 15;          // 0..15  (8-float chunk)
        const float* gsrc = A + ((size_t)blockIdx.x * 16 + r) * DIM + c * 8;
        float*       ldst = &lA[r][c * 8];
#if defined(USE_ASYNC_LDS)
        // one address setup, two B128 transfers; imm offset applies to both
        // the global and LDS sides per the VGLOBAL async encoding.
        as1_v4i g = (as1_v4i)(as1_vptr)(float*)gsrc;
        as3_v4i l = (as3_v4i)(as3_vptr)ldst;
        __builtin_amdgcn_global_load_async_to_lds_b128(g, l, 0, 0);
        __builtin_amdgcn_global_load_async_to_lds_b128(g, l, 16, 0);
        async_wait0();
#else
        const float4* g4 = (const float4*)gsrc;
        float4 v0 = g4[0];
        float4 v1 = g4[1];
        ((float4*)ldst)[0] = v0;
        ((float4*)ldst)[1] = v1;
#endif
    }
    __syncthreads();

    const int lane = threadIdx.x & 31;
    const int wave = threadIdx.x >> 5;
    const int half = lane >> 4;
    const int l15  = lane & 15;
    const int colB = wave * 16 + l15;            // col this lane supplies for B

    v8f acc = {};
#pragma unroll
    for (int kt = 0; kt < 4; ++kt) {
        const int kbase = kt * 32 + half * 8;
        v16h a, b;
#pragma unroll
        for (int j = 0; j < 8; ++j) {
            const int k0 = kbase + j;            // K in [half*8, half*8+8)
            const int k1 = kbase + 16 + j;       // K in [16+half*8, 16+half*8+8)
            a[j]     = (_Float16)lA[l15][k0];
            a[j + 8] = (_Float16)lA[l15][k1];
            b[j]     = (_Float16)B[k0 * DIM + colB];
            b[j + 8] = (_Float16)B[k1 * DIM + colB];
        }
        acc = __builtin_amdgcn_wmma_f32_16x16x32_f16(
            /*neg_a=*/false, a, /*neg_b=*/false, b,
            /*c_mod=*/(short)0, acc, /*reuse_a=*/false, /*reuse_b=*/false);
    }
#pragma unroll
    for (int r = 0; r < 8; ++r)
        C[((size_t)blockIdx.x * 16 + r + 8 * half) * DIM + wave * 16 + l15] = acc[r];
}

// ---------------------------------------------------------------------------
// Per-node attention logits: one wave per node, wave32 shuffle reduction.
// a_src[n][h] = dot(hfeat[n, h*32 : h*32+32], att_src[h]);  same for dst.
// ---------------------------------------------------------------------------
__global__ __launch_bounds__(256) void attn_logits(const float* __restrict__ hfeat,
                                                   const float* __restrict__ att_src,
                                                   const float* __restrict__ att_dst,
                                                   float* __restrict__ a_src,
                                                   float* __restrict__ a_dst) {
    const int lane = threadIdx.x & 31;
    const int node = blockIdx.x * 8 + (threadIdx.x >> 5);
    if (node >= N_NODES) return;
    float ps[HEADS], pd[HEADS];
#pragma unroll
    for (int k = 0; k < HEADS; ++k) {
        float v = hfeat[node * DIM + k * D_HEAD + lane];
        ps[k] = v * att_src[k * D_HEAD + lane];
        pd[k] = v * att_dst[k * D_HEAD + lane];
    }
#pragma unroll
    for (int k = 0; k < HEADS; ++k) {
#pragma unroll
        for (int off = 16; off > 0; off >>= 1) {
            ps[k] += __shfl_down(ps[k], off, 32);
            pd[k] += __shfl_down(pd[k], off, 32);
        }
    }
    if (lane == 0) {
#pragma unroll
        for (int k = 0; k < HEADS; ++k) {
            a_src[node * HEADS + k] = ps[k];
            a_dst[node * HEADS + k] = pd[k];
        }
    }
}

// ---------------------------------------------------------------------------
// Workspace init (ws is poisoned by harness; must re-init every call).
// ---------------------------------------------------------------------------
__global__ void init_ws(float* __restrict__ emax, float* __restrict__ denom,
                        float* __restrict__ colsum, float* __restrict__ colssq) {
    int tid = blockIdx.x * blockDim.x + threadIdx.x;
    if (tid < N_NODES * HEADS) {
        emax[tid]  = -3.402823466e+38f;
        denom[tid] = 0.0f;
    }
    if (tid < DIM) {
        colsum[tid] = 0.0f;
        colssq[tid] = 0.0f;
    }
}

// Monotone float atomic max via signed/unsigned integer punning
// (works for all finite values when the slot is initialized to -FLT_MAX).
__device__ __forceinline__ void atomicMaxFloat(float* addr, float val) {
    if (val >= 0.0f)
        atomicMax((int*)addr, __float_as_int(val));
    else
        atomicMin((unsigned int*)addr, __float_as_uint(val));
}

// ---------------------------------------------------------------------------
// Edge pass 1: leaky-relu logit, stash it, segment-max into emax[dst].
// ---------------------------------------------------------------------------
__global__ void edge_logit_max(const int* __restrict__ src, const int* __restrict__ dst,
                               const float* __restrict__ a_src, const float* __restrict__ a_dst,
                               float* __restrict__ elog, float* __restrict__ emax) {
    int tid = blockIdx.x * blockDim.x + threadIdx.x;
    if (tid >= N_EDGES * HEADS) return;
    int e = tid >> 2, hh = tid & 3;
    int s = src[e], d = dst[e];
    float l = a_src[s * HEADS + hh] + a_dst[d * HEADS + hh];
    l = (l > 0.0f) ? l : NEG_SLOPE * l;
    elog[tid] = l;
    atomicMaxFloat(&emax[d * HEADS + hh], l);
}

// ---------------------------------------------------------------------------
// Edge pass 2: e_exp = exp(l - emax[dst]); stash; segment-sum into denom[dst].
// ---------------------------------------------------------------------------
__global__ void edge_expsum(const int* __restrict__ dst,
                            const float* __restrict__ emax,
                            float* __restrict__ elog, float* __restrict__ denom) {
    int tid = blockIdx.x * blockDim.x + threadIdx.x;
    if (tid >= N_EDGES * HEADS) return;
    int e = tid >> 2, hh = tid & 3;
    int d = dst[e];
    float ex = expf(elog[tid] - emax[d * HEADS + hh]);
    elog[tid] = ex;
    atomicAdd(&denom[d * HEADS + hh], ex);
}

// ---------------------------------------------------------------------------
// Edge pass 3: acc[dst][k] += h[src][k] * alpha.  Thread = (edge, dim); the
// 128 threads of one edge are contiguous -> coalesced gather + atomics.
// ---------------------------------------------------------------------------
__global__ void edge_scatter(const int* __restrict__ src, const int* __restrict__ dst,
                             const float* __restrict__ hfeat, const float* __restrict__ eexp,
                             const float* __restrict__ denom, float* __restrict__ acc) {
    int tid = blockIdx.x * blockDim.x + threadIdx.x;
    if (tid >= N_EDGES * DIM) return;
    int e = tid >> 7, k = tid & 127, hh = k >> 5;
    int s = src[e], d = dst[e];
    float alpha = eexp[e * HEADS + hh] / (denom[d * HEADS + hh] + 1e-16f);
    atomicAdd(&acc[d * DIM + k], hfeat[s * DIM + k] * alpha);
}

// ---------------------------------------------------------------------------
// BatchNorm column stats: thread = column (coalesced row reads), 256 rows per
// block, atomic partial reduction into colsum/colssq.
// ---------------------------------------------------------------------------
__global__ __launch_bounds__(128) void bn_partial(const float* __restrict__ acc,
                                                  float* __restrict__ colsum,
                                                  float* __restrict__ colssq) {
    const int c  = threadIdx.x;
    const int r0 = blockIdx.x * 256;
    float s = 0.0f, q = 0.0f;
    for (int r = r0; r < r0 + 256 && r < N_NODES; ++r) {
        float v = acc[r * DIM + c];
        s += v;
        q += v * v;
    }
    atomicAdd(&colsum[c], s);
    atomicAdd(&colssq[c], q);
}

__global__ void bn_apply(const float* __restrict__ acc,
                         const float* __restrict__ colsum, const float* __restrict__ colssq,
                         const float* __restrict__ gamma, const float* __restrict__ beta,
                         float* __restrict__ out) {
    int tid = blockIdx.x * blockDim.x + threadIdx.x;
    if (tid >= N_NODES * DIM) return;
    int c = tid & 127;
    const float inv_n = 1.0f / (float)N_NODES;
    float mu  = colsum[c] * inv_n;
    float var = colssq[c] * inv_n - mu * mu;
    float y = (acc[tid] - mu) * rsqrtf(var + BN_EPS) * gamma[c] + beta[c];
    out[tid] = fmaxf(y, 0.0f);
}

// ---------------------------------------------------------------------------
extern "C" void kernel_launch(void* const* d_in, const int* in_sizes, int n_in,
                              void* d_out, int out_size, void* d_ws, size_t ws_size,
                              hipStream_t stream) {
    const float* x       = (const float*)d_in[0];   // [N, 128]
    const float* W       = (const float*)d_in[1];   // [128, 128]
    const float* att_src = (const float*)d_in[2];   // [4, 32]
    const float* att_dst = (const float*)d_in[3];   // [4, 32]
    const float* W_lin   = (const float*)d_in[4];   // [128, 128]
    const float* gamma   = (const float*)d_in[5];   // [128]
    const float* beta    = (const float*)d_in[6];   // [128]
    const int*   eidx    = (const int*)d_in[7];     // [2, E]
    const int*   src = eidx;
    const int*   dst = eidx + N_EDGES;

    // Workspace carve-up (floats)
    float* ws = (float*)d_ws;
    float* h      = ws;                         ws += (size_t)N_NODES * DIM;   // 12.8M
    float* acc    = ws;                         ws += (size_t)N_NODES * DIM;   // 12.8M
    float* a_src  = ws;                         ws += (size_t)N_NODES * HEADS; // 0.4M
    float* a_dst  = ws;                         ws += (size_t)N_NODES * HEADS;
    float* emax   = ws;                         ws += (size_t)N_NODES * HEADS;
    float* denom  = ws;                         ws += (size_t)N_NODES * HEADS;
    float* elog   = ws;                         ws += (size_t)N_EDGES * HEADS; // 6.4M
    float* colsum = ws;                         ws += DIM;
    float* colssq = ws;                         ws += DIM;

    float* out = (float*)d_out;

    // 1) h = x @ W          (WMMA)
    gemm128_wmma<<<N_NODES / 16, 256, 0, stream>>>(x, W, h);
    // 2) acc = x @ W_lin    (WMMA; acc is also the conv scatter target)
    gemm128_wmma<<<N_NODES / 16, 256, 0, stream>>>(x, W_lin, acc);
    // 3) per-node attention logits
    attn_logits<<<(N_NODES + 7) / 8, 256, 0, stream>>>(h, att_src, att_dst, a_src, a_dst);
    // 4) init segment-softmax state + BN partials
    init_ws<<<(N_NODES * HEADS + 255) / 256, 256, 0, stream>>>(emax, denom, colsum, colssq);
    // 5) segment max
    edge_logit_max<<<(N_EDGES * HEADS + 255) / 256, 256, 0, stream>>>(src, dst, a_src, a_dst, elog, emax);
    // 6) exp + segment sum
    edge_expsum<<<(N_EDGES * HEADS + 255) / 256, 256, 0, stream>>>(dst, emax, elog, denom);
    // 7) weighted scatter-add of messages
    edge_scatter<<<(N_EDGES * DIM + 255) / 256, 256, 0, stream>>>(src, dst, h, elog, denom, acc);
    // 8) BN stats + apply + ReLU
    bn_partial<<<(N_NODES + 255) / 256, 128, 0, stream>>>(acc, colsum, colssq);
    bn_apply<<<(N_NODES * DIM + 255) / 256, 256, 0, stream>>>(acc, colsum, colssq, gamma, beta, out);
}
